// RelativePropagation_11192684773832
// MI455X (gfx1250) — matrix operand order repaired and checked by reading, not compiled, not run on hardware
//
#include <hip/hip_runtime.h>

typedef __attribute__((ext_vector_type(16))) __bf16 v16bf;
typedef __attribute__((ext_vector_type(8)))  float  v8f;

#define Bdim 2
#define Cdim 64
#define Hdim 192
#define Wdim 384
#define Kpts 8
#define HWSZ (Hdim * Wdim)
#define PIX_PER_BLOCK 128
#define LPAD 17   // LDS row pitch (floats) — kills bank conflicts on per-pixel reads

__global__ __launch_bounds__(256)
void deform_fused_kernel(const float* __restrict__ feat,
                         const unsigned char* __restrict__ mask,
                         const float* __restrict__ Wo,
                         const float* __restrict__ bo,
                         const float* __restrict__ Ww,
                         const float* __restrict__ bw,
                         float* __restrict__ out) {
  __shared__ float offs_lds[PIX_PER_BLOCK][LPAD];    // per-pixel 2K offsets
  __shared__ float logit_lds[PIX_PER_BLOCK][LPAD];   // per-pixel K logits (8 used of 16)

  const int tid  = threadIdx.x;
  const int lane = tid & 31;
  const int wave = tid >> 5;
  const int hi   = lane >> 4;   // half-wave select (ISA A/B layouts)
  const int mrow = lane & 15;   // A-matrix row (pixel)
  const int n    = lane & 15;   // B/C-matrix column (output channel)

  const long blockBase = (long)blockIdx.x * PIX_PER_BLOCK;

  // ---------------- Phase 1: WMMA 1x1 convs ----------------
  // GEMM: [16 pixels x 64 ch] x [64 ch x (16 offs | 8 logits + 8 zero)]
  {
    const long pixflat = blockBase + (long)(wave * 16 + mrow);
    const int b  = (int)(pixflat / HWSZ);
    const int hw = (int)(pixflat % HWSZ);
    const float* fbase = feat + (long)b * Cdim * HWSZ + hw;

    // A-matrix bf16 16x32 layout: element e -> channel hi*16 + e
    v16bf a0, a1;
#pragma unroll
    for (int e = 0; e < 16; ++e) {
      const int ch = hi * 16 + e;
      a0[e] = (__bf16)fbase[(long)ch * HWSZ];
      a1[e] = (__bf16)fbase[(long)(ch + 32) * HWSZ];
    }

    // B-matrix bf16 32x16 layout: lane = N, element e -> K = hi*16 + e (per k-block).
    // Branch-free: clamp row index, scale by 0/1 predicate (keeps EXEC untouched).
    const float sel = (n < 8) ? 1.f : 0.f;
    const int   nw  = n & 7;
    v16bf b00, b01, b10, b11;   // b<Ntile><kblock>
#pragma unroll
    for (int e = 0; e < 16; ++e) {
      const int ch = hi * 16 + e;
      b00[e] = (__bf16)Wo[n * Cdim + ch];
      b01[e] = (__bf16)Wo[n * Cdim + ch + 32];
      b10[e] = (__bf16)(sel * Ww[nw * Cdim + ch]);
      b11[e] = (__bf16)(sel * Ww[nw * Cdim + ch + 32]);
    }

    v8f acc0 = {}, acc1 = {};
    acc0 = __builtin_amdgcn_wmma_f32_16x16x32_bf16(false, a0, false, b00, (short)0, acc0, false, false);
    acc0 = __builtin_amdgcn_wmma_f32_16x16x32_bf16(false, a1, false, b01, (short)0, acc0, false, false);
    acc1 = __builtin_amdgcn_wmma_f32_16x16x32_bf16(false, a0, false, b10, (short)0, acc1, false, false);
    acc1 = __builtin_amdgcn_wmma_f32_16x16x32_bf16(false, a1, false, b11, (short)0, acc1, false, false);

    // C/D layout: lane = N, VGPR r -> pixel M = r + hi*8
    const float bo_n = bo[n];
    const float bw_n = sel * bw[nw];
#pragma unroll
    for (int r = 0; r < 8; ++r) {
      const int p = wave * 16 + r + hi * 8;
      offs_lds[p][n]  = acc0[r] + bo_n;
      logit_lds[p][n] = acc1[r] + bw_n;   // unconditional; cols 8..15 unused
    }
  }
  __syncthreads();

  // ---------------- Phase 2: softmax + bilinear gather ----------------
  // Waves 0-3: channels 0..31 of pixels 0..127; waves 4-7: channels 32..63.
  // Adjacent lanes = adjacent pixels at same channel -> coalesced center I/O.
  {
    const int pixl = tid & 127;
    const int half = tid >> 7;
    const long pixflat = blockBase + pixl;
    const int b  = (int)(pixflat / HWSZ);
    const int hw = (int)(pixflat % HWSZ);
    const int h  = hw / Wdim;
    const int w  = hw % Wdim;

    // softmax over K=8 (duplicated across the 2 halves; cheap)
    float lg[Kpts];
    float mx = -1e30f;
#pragma unroll
    for (int k = 0; k < Kpts; ++k) { lg[k] = logit_lds[pixl][k]; mx = fmaxf(mx, lg[k]); }
    float sum = 0.f;
#pragma unroll
    for (int k = 0; k < Kpts; ++k) { lg[k] = __expf(lg[k] - mx); sum += lg[k]; }
    const float inv = 1.f / sum;

    float wk[Kpts], px[Kpts], py[Kpts];
#pragma unroll
    for (int k = 0; k < Kpts; ++k) {
      wk[k] = lg[k] * inv;
      px[k] = (float)w + offs_lds[pixl][2 * k];       // align_corners=True pixel coords
      py[k] = (float)h + offs_lds[pixl][2 * k + 1];
    }

    const float mval = mask[(long)b * HWSZ + hw] ? 1.f : 0.f;
    const float* fb0 = feat + (long)b * Cdim * HWSZ;
    float*       ob0 = out  + (long)b * Cdim * HWSZ;

#pragma unroll
    for (int chunk = 0; chunk < 2; ++chunk) {
      const int cbase = half * 32 + chunk * 16;
      float res[16];
#pragma unroll
      for (int c = 0; c < 16; ++c) res[c] = 0.f;

      for (int k = 0; k < Kpts; ++k) {
        const float x0f = floorf(px[k]);
        const float y0f = floorf(py[k]);
        const float wx = px[k] - x0f;
        const float wy = py[k] - y0f;
        const int x0 = (int)x0f, y0 = (int)y0f;
        const int x1 = x0 + 1,   y1 = y0 + 1;
        // zero-padding: weight *= in-bounds; index clipped (matches reference)
        const float vx0 = (x0 >= 0 && x0 < Wdim) ? 1.f : 0.f;
        const float vx1 = (x1 >= 0 && x1 < Wdim) ? 1.f : 0.f;
        const float vy0 = (y0 >= 0 && y0 < Hdim) ? 1.f : 0.f;
        const float vy1 = (y1 >= 0 && y1 < Hdim) ? 1.f : 0.f;
        const int x0c = min(max(x0, 0), Wdim - 1);
        const int x1c = min(max(x1, 0), Wdim - 1);
        const int y0c = min(max(y0, 0), Hdim - 1);
        const int y1c = min(max(y1, 0), Hdim - 1);
        const float w00 = wk[k] * (1.f - wx) * (1.f - wy) * vx0 * vy0;
        const float w10 = wk[k] * wx * (1.f - wy) * vx1 * vy0;
        const float w01 = wk[k] * (1.f - wx) * wy * vx0 * vy1;
        const float w11 = wk[k] * wx * wy * vx1 * vy1;
        const int i00 = y0c * Wdim + x0c;
        const int i10 = y0c * Wdim + x1c;
        const int i01 = y1c * Wdim + x0c;
        const int i11 = y1c * Wdim + x1c;
        const float* f = fb0 + (long)cbase * HWSZ;
#pragma unroll
        for (int c = 0; c < 16; ++c) {
          const float* fc = f + (long)c * HWSZ;   // gathers hit the 192 MB L2 (feat = 37.7 MB)
          res[c] += w00 * fc[i00] + w10 * fc[i10] + w01 * fc[i01] + w11 * fc[i11];
        }
      }
#pragma unroll
      for (int c = 0; c < 16; ++c) {
        const long idx = (long)(cbase + c) * HWSZ + hw;
        ob0[idx] = fb0[idx] + res[c] * mval;
      }
    }
  }
}

extern "C" void kernel_launch(void* const* d_in, const int* in_sizes, int n_in,
                              void* d_out, int out_size, void* d_ws, size_t ws_size,
                              hipStream_t stream) {
  const float*         feat = (const float*)d_in[0];
  const unsigned char* mask = (const unsigned char*)d_in[1];  // jnp bool_ = 1 byte
  const float*         Wo   = (const float*)d_in[2];
  const float*         bo   = (const float*)d_in[3];
  const float*         Ww   = (const float*)d_in[4];
  const float*         bw   = (const float*)d_in[5];
  float* out = (float*)d_out;

  const int nblocks = (Bdim * Hdim * Wdim) / PIX_PER_BLOCK;  // 1152
  deform_fused_kernel<<<nblocks, 256, 0, stream>>>(feat, mask, Wo, bo, Ww, bw, out);
}
